// MultiHeadSelfAttention_63299228008591
// MI455X (gfx1250) — matrix operand order repaired and checked by reading, not compiled, not run on hardware
//
#include <hip/hip_runtime.h>
#include <hip/hip_bf16.h>

#define B_  8
#define S_  1024
#define D_  768
#define H_  12
#define DK_ 64

typedef __attribute__((ext_vector_type(16))) __bf16 v16bf;
typedef __attribute__((ext_vector_type(8)))  float  v8f;
typedef int v4i_ __attribute__((vector_size(16)));

union BF16x16 { unsigned short u[16]; uint4 q[2]; v16bf v; };
union BF16x8  { unsigned short u[8];  uint4 q; };

__device__ __forceinline__ unsigned short f2bf(float f) {
  union { float f; unsigned int u; } x; x.f = f;
  unsigned int u = x.u;
  return (unsigned short)((u + 0x7FFFu + ((u >> 16) & 1u)) >> 16);
}
__device__ __forceinline__ float bf2f(unsigned short h) {
  union { unsigned int u; float f; } x; x.u = ((unsigned int)h) << 16;
  return x.f;
}
__device__ __forceinline__ v8f wmma_bf16(v16bf a, v16bf b, v8f c) {
  return __builtin_amdgcn_wmma_f32_16x16x32_bf16(false, a, false, b, (short)0, c, false, false);
}
// Two ds_load_b128 per fragment (p must be 16B aligned).
__device__ __forceinline__ void load_frag(BF16x16& f, const unsigned short* p) {
  f.q[0] = *(const uint4*)p;
  f.q[1] = *(const uint4*)(p + 8);
}
// A-tile K-block permutation (blocks of 8): 0,1,2,3 -> 0,2,1,3
__device__ __forceinline__ int ablk(int b) { return (b == 1) ? 2 : (b == 2) ? 1 : b; }

// ---- async global->LDS 16B staging (ASYNCcnt path), with sync fallback ----
#if __has_builtin(__builtin_amdgcn_global_load_async_to_lds_b128)
#define HAVE_ASYNC_LDS 1
#endif

__device__ __forceinline__ void stage16(const unsigned short* g, unsigned short* l) {
#ifdef HAVE_ASYNC_LDS
  __builtin_amdgcn_global_load_async_to_lds_b128(
      (__attribute__((address_space(1))) v4i_*)g,
      (__attribute__((address_space(3))) v4i_*)l, 0, 0);
#else
  *(uint4*)l = *(const uint4*)g;
#endif
}
__device__ __forceinline__ void stage_fence() {
#ifdef HAVE_ASYNC_LDS
#if __has_builtin(__builtin_amdgcn_s_wait_asynccnt)
  __builtin_amdgcn_s_wait_asynccnt(0);
#else
  asm volatile("s_wait_asynccnt 0" ::: "memory");
#endif
#endif
}

// ---------------------------------------------------------------------------
// Kernel 0: bulk f32 -> bf16 conversion (bandwidth-bound, one pass)
// ---------------------------------------------------------------------------
__global__ __launch_bounds__(256) void cvt_bf16_kernel(
    const float* __restrict__ src, unsigned short* __restrict__ dst, int n8)
{
  int i = blockIdx.x * 256 + threadIdx.x;
  if (i >= n8) return;
  const float4* s = (const float4*)src + (size_t)i * 2;
  float4 f0 = s[0], f1 = s[1];
  BF16x8 t;
  t.u[0] = f2bf(f0.x); t.u[1] = f2bf(f0.y); t.u[2] = f2bf(f0.z); t.u[3] = f2bf(f0.w);
  t.u[4] = f2bf(f1.x); t.u[5] = f2bf(f1.y); t.u[6] = f2bf(f1.z); t.u[7] = f2bf(f1.w);
  ((uint4*)dst)[i] = t.q;
}

// ---------------------------------------------------------------------------
// Kernel 1: Q/K/V projections. Xb[8192,768](bf16) @ Wb^T(bf16).
// Q,K -> bf16 [B,H,S,DK]; V -> bf16 [B,H,DK,S] (transposed for attn@V B-frags)
// ---------------------------------------------------------------------------
__global__ __launch_bounds__(256) void proj_qkv_kernel(
    const unsigned short* __restrict__ Xq, const unsigned short* __restrict__ Xk,
    const unsigned short* __restrict__ Xv,
    const unsigned short* __restrict__ Wqb, const unsigned short* __restrict__ Wkb,
    const unsigned short* __restrict__ Wvb,
    unsigned short* __restrict__ Qh, unsigned short* __restrict__ Kh,
    unsigned short* __restrict__ Vh)
{
  const int z = blockIdx.z;
  const unsigned short* X = (z == 0) ? Xq  : (z == 1) ? Xk  : Xv;
  const unsigned short* W = (z == 0) ? Wqb : (z == 1) ? Wkb : Wvb;
  unsigned short* dst = (z == 0) ? Qh : (z == 1) ? Kh : Vh;

  const int m0 = blockIdx.x * 128;
  const int n0 = blockIdx.y * 128;

  __shared__ unsigned short Al[128][40];   // swizzled A-fragment layout
  __shared__ unsigned short Bl[128][40];   // Bl[j][c] (frag-major)

  const int tid  = threadIdx.x;
  const int lane = tid & 31;
  const int wv   = tid >> 5;
  const int grp  = lane >> 4;
  const int lr   = lane & 15;

  v8f ct[8];
#pragma unroll
  for (int t = 0; t < 8; ++t)
#pragma unroll
    for (int i = 0; i < 8; ++i) ct[t][i] = 0.0f;

  for (int k0 = 0; k0 < D_; k0 += 32) {
    {
      int row = tid >> 1, sg = tid & 1;
      const unsigned short* s = X + (size_t)(m0 + row) * D_ + k0 + sg * 16;
      stage16(s,     &Al[row][sg * 8]);       // K-blocks 0->0, 2->1
      stage16(s + 8, &Al[row][16 + sg * 8]);  // K-blocks 1->2, 3->3
    }
    {
      int jloc = tid >> 1, cb = (tid & 1) * 16;
      const unsigned short* s = W + (size_t)(n0 + jloc) * D_ + k0 + cb;
      stage16(s,     &Bl[jloc][cb / 2]);
      stage16(s + 8, &Bl[jloc][16 + cb / 2]);
    }
    stage_fence();
    __syncthreads();

    BF16x16 a, bf[8];
    load_frag(a, &Al[wv * 16 + lr][grp * 16]);
#pragma unroll
    for (int t = 0; t < 8; ++t) load_frag(bf[t], &Bl[t * 16 + lr][grp * 16]);
#pragma unroll
    for (int t = 0; t < 8; ++t) ct[t] = wmma_bf16(a.v, bf[t].v, ct[t]);
    __syncthreads();
  }

#pragma unroll
  for (int t = 0; t < 8; ++t)
#pragma unroll
    for (int i = 0; i < 8; ++i) {
      int m = m0 + wv * 16 + i + 8 * grp;
      int j = n0 + t * 16 + lr;
      int bb = m >> 10, s = m & 1023;
      int h = j >> 6, d = j & 63;
      unsigned short val = f2bf(ct[t][i]);
      if (z == 2)
        dst[(((size_t)bb * H_ + h) * DK_ + d) * S_ + s] = val;   // V transposed
      else
        dst[(((size_t)bb * H_ + h) * S_ + s) * DK_ + d] = val;
    }
}

// ---------------------------------------------------------------------------
// Kernel 2: energy = QK^T/8, mask, softmax -> bf16 attn [B,H,S,S]
// ---------------------------------------------------------------------------
__global__ __launch_bounds__(256) void attn_softmax_kernel(
    const unsigned short* __restrict__ Qh, const unsigned short* __restrict__ Kh,
    const unsigned char* __restrict__ mask, unsigned short* __restrict__ attnw)
{
  const int q0 = blockIdx.x * 16;
  const int h  = blockIdx.y;
  const int b  = blockIdx.z;

  __shared__ unsigned short Ql[16][72];    // swizzled A layout (two 32-K halves)
  __shared__ unsigned short Kl[128][72];   // [kcol][dk], dk K-blocks swizzled like Ql (frag-major B)
  __shared__ float redbuf[16][8];
  __shared__ float rowred[16];

  const int tid  = threadIdx.x;
  const int lane = tid & 31;
  const int wv   = tid >> 5;
  const int grp  = lane >> 4;
  const int lr   = lane & 15;

  const size_t headbase = ((size_t)b * H_ + h) * S_;

  if (tid < 128) {
    int row = tid >> 3, seg = tid & 7;
    int half = seg >> 2, bi = seg & 3;
    stage16(Qh + (headbase + q0 + row) * DK_ + seg * 8,
            &Ql[row][half * 32 + ablk(bi) * 8]);
  }
  stage_fence();
  __syncthreads();

  BF16x16 a0, a1;
  load_frag(a0, &Ql[lr][grp * 16]);
  load_frag(a1, &Ql[lr][32 + grp * 16]);

  v8f ct[8];
#pragma unroll
  for (int c = 0; c < 8; ++c)
#pragma unroll
    for (int i = 0; i < 8; ++i) ct[c][i] = 0.0f;

#pragma unroll 1
  for (int c = 0; c < 8; ++c) {
    {
      int row = tid >> 1, seg = (tid & 1) * 32;
      const unsigned short* s = Kh + (headbase + c * 128 + row) * DK_ + seg;
#pragma unroll
      for (int u = 0; u < 4; ++u)
        stage16(s + u * 8, &Kl[row][seg + ablk(u) * 8]);
    }
    stage_fence();
    __syncthreads();
    BF16x16 b0, b1;
    int col = wv * 16 + lr;
    load_frag(b0, &Kl[col][grp * 16]);
    load_frag(b1, &Kl[col][32 + grp * 16]);
    v8f cc = wmma_bf16(a0.v, b0.v, ct[c]);
    ct[c] = wmma_bf16(a1.v, b1.v, cc);
    __syncthreads();
  }

  const unsigned char* mb = mask + (size_t)b * S_ * S_;
  float lmax[8];
#pragma unroll
  for (int i = 0; i < 8; ++i) lmax[i] = -INFINITY;
#pragma unroll
  for (int c = 0; c < 8; ++c)
#pragma unroll
    for (int i = 0; i < 8; ++i) {
      int qq = q0 + i + 8 * grp;
      int kk = c * 128 + wv * 16 + lr;
      float x = ct[c][i] * 0.125f;
      if (mb[(size_t)qq * S_ + kk]) x = -INFINITY;
      ct[c][i] = x;
      lmax[i] = fmaxf(lmax[i], x);
    }
#pragma unroll
  for (int off = 8; off; off >>= 1)
#pragma unroll
    for (int i = 0; i < 8; ++i) lmax[i] = fmaxf(lmax[i], __shfl_xor(lmax[i], off, 32));
  if (lr == 0)
#pragma unroll
    for (int i = 0; i < 8; ++i) redbuf[i + 8 * grp][wv] = lmax[i];
  __syncthreads();
  if (tid < 16) {
    float m = -INFINITY;
#pragma unroll
    for (int w = 0; w < 8; ++w) m = fmaxf(m, redbuf[tid][w]);
    rowred[tid] = m;
  }
  __syncthreads();

  float rr[8];
#pragma unroll
  for (int i = 0; i < 8; ++i) rr[i] = rowred[i + 8 * grp];
  __syncthreads();

  float lsum[8];
#pragma unroll
  for (int i = 0; i < 8; ++i) lsum[i] = 0.0f;
#pragma unroll
  for (int c = 0; c < 8; ++c)
#pragma unroll
    for (int i = 0; i < 8; ++i) {
      float p = __expf(ct[c][i] - rr[i]);
      ct[c][i] = p;
      lsum[i] += p;
    }
#pragma unroll
  for (int off = 8; off; off >>= 1)
#pragma unroll
    for (int i = 0; i < 8; ++i) lsum[i] += __shfl_xor(lsum[i], off, 32);
  if (lr == 0)
#pragma unroll
    for (int i = 0; i < 8; ++i) redbuf[i + 8 * grp][wv] = lsum[i];
  __syncthreads();
  if (tid < 16) {
    float s = 0.0f;
#pragma unroll
    for (int w = 0; w < 8; ++w) s += redbuf[tid][w];
    rowred[tid] = 1.0f / s;
  }
  __syncthreads();

#pragma unroll
  for (int i = 0; i < 8; ++i) rr[i] = rowred[i + 8 * grp];   // inv-sum
#pragma unroll
  for (int c = 0; c < 8; ++c)
#pragma unroll
    for (int i = 0; i < 8; ++i) {
      int qq = q0 + i + 8 * grp;
      int kk = c * 128 + wv * 16 + lr;
      attnw[(headbase + qq) * S_ + kk] = f2bf(ct[c][i] * rr[i]);
    }
}

// ---------------------------------------------------------------------------
// Kernel 3: re-attention (head mix + BN affine) fused with attn2 @ V.
// 384 threads = 12 waves; wave g -> head g's 16x64 output tile.
// ---------------------------------------------------------------------------
__global__ __launch_bounds__(384) void reattn_av_kernel(
    const unsigned short* __restrict__ attnw, const unsigned short* __restrict__ Vh,
    const float* __restrict__ conv_w, const float* __restrict__ conv_b,
    const float* __restrict__ bn_gamma, const float* __restrict__ bn_beta,
    const float* __restrict__ bn_mean, const float* __restrict__ bn_var,
    unsigned short* __restrict__ O)
{
  extern __shared__ char smem[];
  float* cw2         = (float*)smem;                    // 144 f32
  float* shiftv      = (float*)(smem + 576);            // 12 f32
  unsigned short* At = (unsigned short*)(smem + 640);   // 12*16*40 = 15360B
  unsigned short* Vt = (unsigned short*)(smem + 16000); // 12*64*40 = 61440B

  const int q0 = blockIdx.x * 16;
  const int b  = blockIdx.y;

  const int tid  = threadIdx.x;
  const int lane = tid & 31;
  const int g    = tid >> 5;
  const int grp  = lane >> 4;
  const int lr   = lane & 15;

  if (tid < 144) {
    int gg = tid / 12, hh = tid % 12;
    float sc = bn_gamma[gg] * rsqrtf(bn_var[gg] + 1e-5f);
    cw2[gg * 12 + hh] = sc * conv_w[gg * 12 + hh];
    if (hh == 0) shiftv[gg] = (conv_b[gg] - bn_mean[gg]) * sc + bn_beta[gg];
  }
  __syncthreads();

  v8f ct[4];
#pragma unroll
  for (int t = 0; t < 4; ++t)
#pragma unroll
    for (int i = 0; i < 8; ++i) ct[t][i] = 0.0f;

#pragma unroll 1
  for (int kc = 0; kc < 32; ++kc) {
    const int k0 = kc * 32;
    // attn tiles: 12 heads x 16 q x 32 k, stored swizzled (768 uint4)
    for (int idx = tid; idx < 768; idx += 384) {
      int hh = idx >> 6, r = idx & 63;
      int qr = r >> 2, bi = r & 3;
      stage16(attnw + (((size_t)b * H_ + hh) * S_ + q0 + qr) * S_ + k0 + bi * 8,
              &At[(hh * 16 + qr) * 40 + ablk(bi) * 8]);
    }
    // V tiles (transposed global [B,H,DK,S]): 12 heads x 64 d x 32 k (3072 uint4)
    for (int idx = tid; idx < 3072; idx += 384) {
      int gg = idx >> 8, r = idx & 255;
      int dd = r >> 2, seg = (r & 3) * 8;
      stage16(Vh + (((size_t)b * H_ + gg) * DK_ + dd) * S_ + k0 + seg,
              &Vt[(gg * 64 + dd) * 40 + ablk(r & 3) * 8]);
    }
    stage_fence();
    __syncthreads();

    // preload V fragments so WMMAs can issue right after mixing completes
    BF16x16 vf[4];
#pragma unroll
    for (int t = 0; t < 4; ++t)
      load_frag(vf[t], &Vt[(g * 64 + t * 16 + lr) * 40 + grp * 16]);

    // mixed A-fragment: attn2[g,q,k] = sum_h cw2[g,h]*attn[h,q,k] + shift
    float acc[16];
    float sh = shiftv[g];
#pragma unroll
    for (int j = 0; j < 16; ++j) acc[j] = sh;
#pragma unroll
    for (int hh = 0; hh < 12; ++hh) {
      BF16x16 t;
      load_frag(t, &At[(hh * 16 + lr) * 40 + grp * 16]);
      float w = cw2[g * 12 + hh];
#pragma unroll
      for (int j = 0; j < 16; ++j) acc[j] += w * bf2f(t.u[j]);
    }
    BF16x16 a;
#pragma unroll
    for (int j = 0; j < 16; ++j) a.u[j] = f2bf(acc[j]);

#pragma unroll
    for (int t = 0; t < 4; ++t) ct[t] = wmma_bf16(a.v, vf[t].v, ct[t]);
    __syncthreads();
  }

#pragma unroll
  for (int t = 0; t < 4; ++t)
#pragma unroll
    for (int i = 0; i < 8; ++i) {
      int qq = q0 + i + 8 * grp;
      int col = g * 64 + t * 16 + lr;
      O[((size_t)b * S_ + qq) * D_ + col] = f2bf(ct[t][i]);
    }
}

// ---------------------------------------------------------------------------
// Kernel 4: output projection  O[8192,768](bf16) @ Wo^T(bf16) + bo -> f32 out
// ---------------------------------------------------------------------------
__global__ __launch_bounds__(256) void out_proj_kernel(
    const unsigned short* __restrict__ Og, const unsigned short* __restrict__ Wob,
    const float* __restrict__ bo, float* __restrict__ out)
{
  const int m0 = blockIdx.x * 128;
  const int n0 = blockIdx.y * 128;

  __shared__ unsigned short Al[128][40];
  __shared__ unsigned short Bl[128][40];

  const int tid  = threadIdx.x;
  const int lane = tid & 31;
  const int wv   = tid >> 5;
  const int grp  = lane >> 4;
  const int lr   = lane & 15;

  v8f ct[8];
#pragma unroll
  for (int t = 0; t < 8; ++t)
#pragma unroll
    for (int i = 0; i < 8; ++i) ct[t][i] = 0.0f;

  for (int k0 = 0; k0 < D_; k0 += 32) {
    {
      int row = tid >> 1, sg = tid & 1;
      const unsigned short* s = Og + (size_t)(m0 + row) * D_ + k0 + sg * 16;
      stage16(s,     &Al[row][sg * 8]);
      stage16(s + 8, &Al[row][16 + sg * 8]);
    }
    {
      int jloc = tid >> 1, cb = (tid & 1) * 16;
      const unsigned short* s = Wob + (size_t)(n0 + jloc) * D_ + k0 + cb;
      stage16(s,     &Bl[jloc][cb / 2]);
      stage16(s + 8, &Bl[jloc][16 + cb / 2]);
    }
    stage_fence();
    __syncthreads();

    BF16x16 a, bf[8];
    load_frag(a, &Al[wv * 16 + lr][grp * 16]);
#pragma unroll
    for (int t = 0; t < 8; ++t) load_frag(bf[t], &Bl[t * 16 + lr][grp * 16]);
#pragma unroll
    for (int t = 0; t < 8; ++t) ct[t] = wmma_bf16(a.v, bf[t].v, ct[t]);
    __syncthreads();
  }

#pragma unroll
  for (int t = 0; t < 8; ++t)
#pragma unroll
    for (int i = 0; i < 8; ++i) {
      int m = m0 + wv * 16 + i + 8 * grp;
      int j = n0 + t * 16 + lr;
      out[(size_t)m * D_ + j] = ct[t][i] + bo[j];
    }
}

// ---------------------------------------------------------------------------
extern "C" void kernel_launch(void* const* d_in, const int* in_sizes, int n_in,
                              void* d_out, int out_size, void* d_ws, size_t ws_size,
                              hipStream_t stream) {
  (void)in_sizes; (void)n_in; (void)out_size; (void)ws_size;
  const float* q  = (const float*)d_in[0];
  const float* k  = (const float*)d_in[1];
  const float* v  = (const float*)d_in[2];
  const unsigned char* mask = (const unsigned char*)d_in[3];
  const float* Wq = (const float*)d_in[4];
  const float* Wk = (const float*)d_in[5];
  const float* Wv = (const float*)d_in[6];
  const float* Wo = (const float*)d_in[7];
  const float* bo = (const float*)d_in[8];
  const float* conv_w   = (const float*)d_in[9];
  const float* conv_b   = (const float*)d_in[10];
  const float* bn_gamma = (const float*)d_in[11];
  const float* bn_beta  = (const float*)d_in[12];
  const float* bn_mean  = (const float*)d_in[13];
  const float* bn_var   = (const float*)d_in[14];

  char* ws = (char*)d_ws;
  const size_t szH = (size_t)B_ * H_ * S_ * DK_ * sizeof(unsigned short); // 12.58 MB
  const size_t szA = (size_t)B_ * H_ * S_ * S_ * sizeof(unsigned short);  // 192 MB
  const size_t szW = (size_t)D_ * D_ * sizeof(unsigned short);            // 1.18 MB
  unsigned short* Qh    = (unsigned short*)(ws);
  unsigned short* Kh    = (unsigned short*)(ws + szH);
  unsigned short* Vh    = (unsigned short*)(ws + 2 * szH);   // [B,H,DK,S]
  unsigned short* O     = (unsigned short*)(ws + 3 * szH);
  unsigned short* attnw = (unsigned short*)(ws + 4 * szH);
  char* p = ws + 4 * szH + szA;
  unsigned short* Xqb = (unsigned short*)(p);            p += szH;
  unsigned short* Xkb = (unsigned short*)(p);            p += szH;
  unsigned short* Xvb = (unsigned short*)(p);            p += szH;
  unsigned short* Wqb = (unsigned short*)(p);            p += szW;
  unsigned short* Wkb = (unsigned short*)(p);            p += szW;
  unsigned short* Wvb = (unsigned short*)(p);            p += szW;
  unsigned short* Wob = (unsigned short*)(p);            p += szW;

  const int nX8 = B_ * S_ * D_ / 8;   // 786432
  const int nW8 = D_ * D_ / 8;        // 73728
  cvt_bf16_kernel<<<dim3((nX8 + 255) / 256), 256, 0, stream>>>(q,  Xqb, nX8);
  cvt_bf16_kernel<<<dim3((nX8 + 255) / 256), 256, 0, stream>>>(k,  Xkb, nX8);
  cvt_bf16_kernel<<<dim3((nX8 + 255) / 256), 256, 0, stream>>>(v,  Xvb, nX8);
  cvt_bf16_kernel<<<dim3((nW8 + 255) / 256), 256, 0, stream>>>(Wq, Wqb, nW8);
  cvt_bf16_kernel<<<dim3((nW8 + 255) / 256), 256, 0, stream>>>(Wk, Wkb, nW8);
  cvt_bf16_kernel<<<dim3((nW8 + 255) / 256), 256, 0, stream>>>(Wv, Wvb, nW8);
  cvt_bf16_kernel<<<dim3((nW8 + 255) / 256), 256, 0, stream>>>(Wo, Wob, nW8);

  proj_qkv_kernel<<<dim3(64, 6, 3), 256, 0, stream>>>(
      Xqb, Xkb, Xvb, Wqb, Wkb, Wvb, Qh, Kh, Vh);
  attn_softmax_kernel<<<dim3(S_ / 16, H_, B_), 256, 0, stream>>>(Qh, Kh, mask, attnw);
  const size_t smem3 = 16000 + (size_t)12 * 64 * 40 * sizeof(unsigned short); // 77440 B
  reattn_av_kernel<<<dim3(S_ / 16, B_), 384, smem3, stream>>>(
      attnw, Vh, conv_w, conv_b, bn_gamma, bn_beta, bn_mean, bn_var, O);
  out_proj_kernel<<<dim3(64, 6), 256, 0, stream>>>(O, Wob, bo, (float*)d_out);
}
